// PagedAttentionBlock_23252952940866
// MI455X (gfx1250) — matrix operand order repaired and checked by reading, not compiled, not run on hardware
//
#include <hip/hip_runtime.h>
#include <hip/hip_bf16.h>

// ---------------- problem constants (match reference) ----------------
constexpr int kE = 2048;   // embed dim
constexpr int kH = 16;     // heads
constexpr int kD = 128;    // head dim
constexpr int kB = 8;      // batch
constexpr int kS = 16;     // new tokens
constexpr int kPS = 16;    // page size
constexpr int kCTX = 4080; // past tokens
constexpr int kBLOCKS = 256;
constexpr int kT = 32768;  // physical slots
constexpr int kSPLIT = 8;  // kv-split waves per (b,h)

typedef __attribute__((ext_vector_type(16))) __bf16 v16bf;
typedef __attribute__((ext_vector_type(8)))  float  v8f;

// ---------------- WMMA fragment loaders (fp32 -> bf16) ----------------
// A 16x32 (MxK), lane<16: row=lane, K=0..15; lane>=16: row=lane-16, K=16..31
__device__ __forceinline__ v16bf load_a_frag(const float* __restrict__ row_ptr, int k0) {
  int hi = (threadIdx.x >> 4) & 1;
  const float* p = row_ptr + k0 + hi * 16;
  v16bf a;
#pragma unroll
  for (int i = 0; i < 8; ++i) a[i] = (__bf16)p[i];
#pragma unroll
  for (int i = 0; i < 8; ++i) a[8 + i] = (__bf16)p[8 + i];
  return a;
}

// B 32x16 (KxN), lane<16: N=lane, K=0..15; lane>=16: N=lane-16, K=16..31 (16 consecutive K)
__device__ __forceinline__ v16bf load_b_frag(const float* __restrict__ col_ptr, int k0) {
  int hi = (threadIdx.x >> 4) & 1;
  const float* p = col_ptr + k0 + hi * 16;
  v16bf b;
#pragma unroll
  for (int i = 0; i < 16; ++i) b[i] = (__bf16)p[i];
  return b;
}

// ---------------- kernel 1: QKV projection (out = hidden @ W^T) ----------------
// grid: 3 mats * 8 mtiles * 128 ntiles = 3072 waves -> 384 blocks x 256 thr
__global__ __launch_bounds__(256) void qkv_proj_kernel(
    const float* __restrict__ hidden, const float* __restrict__ wq,
    const float* __restrict__ wk, const float* __restrict__ wv,
    float* __restrict__ q_ws, float* __restrict__ k_new, float* __restrict__ v_new) {
  int wave = blockIdx.x * (blockDim.x >> 5) + (threadIdx.x >> 5);
  int lane = threadIdx.x & 31;
  int ln = lane & 15, hi = lane >> 4;
  int mat = wave >> 10;          // 0..2
  int rem = wave & 1023;
  int m0 = (rem >> 7) * 16;      // row tile in [0,128)
  int n0 = (rem & 127) * 16;     // col tile in [0,2048)
  const float* W = (mat == 0) ? wq : (mat == 1) ? wk : wv;
  float* dst = (mat == 0) ? q_ws : (mat == 1) ? k_new : v_new;

  const float* arow = hidden + (size_t)(m0 + ln) * kE;
  const float* wrow = W + (size_t)(n0 + ln) * kE;   // B[k][n] = W[n][k]
  v8f acc = {};
  for (int k0 = 0; k0 < kE; k0 += 32) {
    v16bf a = load_a_frag(arow, k0);
    v16bf b = load_b_frag(wrow, k0);
    acc = __builtin_amdgcn_wmma_f32_16x16x32_bf16(false, a, false, b, (short)0, acc, false, false);
  }
  // scatter-store into [B,H,S,D]
  int e = n0 + ln, hh = e >> 7, d = e & 127;
#pragma unroll
  for (int r = 0; r < 8; ++r) {
    int row = m0 + r + 8 * hi;     // b*S + s
    int b_ = row >> 4, s = row & 15;
    dst[(((size_t)(b_ * kH + hh)) * kS + s) * kD + d] = acc[r];
  }
}

// ---------------- kernel 2: flash-decoding attention (split KV) ----------------
// grid: B*H = 128 blocks, 256 threads (8 waves); wave w handles chunks w, w+8, ...
__global__ __launch_bounds__(256) void attn_kernel(
    const float* __restrict__ q_ws, const float* __restrict__ k_new,
    const float* __restrict__ v_new, const float* __restrict__ k_cache,
    const float* __restrict__ v_cache, const int* __restrict__ page_table,
    float* __restrict__ part_acc, float* __restrict__ part_ml) {
  __shared__ float pbuf[kSPLIT][16][32];   // per-wave P staging (C-layout -> A-layout)

  int bh = blockIdx.x;                 // b*H + h
  int b = bh >> 4, h = bh & 15;
  int wv_ = threadIdx.x >> 5;
  int lane = threadIdx.x & 31;
  int ln = lane & 15, hi = lane >> 4;
  const float scale = 0.08838834764831845f;   // 1/sqrt(128)

  // Q fragments (scale folded in), reused across all chunks
  const float* qrow = q_ws + ((size_t)bh * kS + ln) * kD;
  v16bf qf[4];
#pragma unroll
  for (int j = 0; j < 4; ++j) {
    const float* p = qrow + j * 32 + hi * 16;
    v16bf a;
#pragma unroll
    for (int i = 0; i < 8; ++i) a[i] = (__bf16)(p[i] * scale);
#pragma unroll
    for (int i = 0; i < 8; ++i) a[8 + i] = (__bf16)(p[8 + i] * scale);
    qf[j] = a;
  }

  v8f acc[8];
#pragma unroll
  for (int n = 0; n < 8; ++n) acc[n] = (v8f){};
  float m_run[8], l_run[8];
#pragma unroll
  for (int r = 0; r < 8; ++r) { m_run[r] = -__builtin_inff(); l_run[r] = 0.f; }

  const float* kbase = k_cache + (size_t)h * kT * kD;
  const float* vbase = v_cache + (size_t)h * kT * kD;
  const float* knb = k_new + (size_t)bh * kS * kD;
  const float* vnb = v_new + (size_t)bh * kS * kD;

  for (int c = wv_; c < 128; c += kSPLIT) {
    int kv0 = c * 32;
    int tok0 = kv0 + ln;         // K-fragment column token, tile 0
    int tok1 = kv0 + 16 + ln;    // tile 1
    int pg0 = page_table[b * kBLOCKS + (kv0 >> 4)];
    int pg1 = page_table[b * kBLOCKS + (kv0 >> 4) + 1];
    const float* kp0 = (tok0 < kCTX) ? (kbase + ((size_t)pg0 * kPS + ln) * kD)
                                     : (knb + (size_t)(tok0 - kCTX) * kD);
    const float* kp1 = (tok1 < kCTX) ? (kbase + ((size_t)pg1 * kPS + ln) * kD)
                                     : (knb + (size_t)(tok1 - kCTX) * kD);
    // prefetch next chunk's K page (global_prefetch_b8)
    int cn = c + kSPLIT;
    if (cn < 128) {
      int pgn = page_table[b * kBLOCKS + (cn << 1)];
      __builtin_prefetch(kbase + ((size_t)pgn * kPS + ln) * kD, 0, 0);
    }

    // scores = Q . K^T  (two 16x16 tiles, K-dim = d, 4 steps of 32)
    v8f c0 = (v8f){}, c1 = (v8f){};
#pragma unroll
    for (int j = 0; j < 4; ++j) {
      int dofs = j * 32 + hi * 16;
      v16bf b0, b1;
#pragma unroll
      for (int i = 0; i < 16; ++i) { b0[i] = (__bf16)kp0[dofs + i]; b1[i] = (__bf16)kp1[dofs + i]; }
      c0 = __builtin_amdgcn_wmma_f32_16x16x32_bf16(false, qf[j], false, b0, (short)0, c0, false, false);
      c1 = __builtin_amdgcn_wmma_f32_16x16x32_bf16(false, qf[j], false, b1, (short)0, c1, false, false);
    }

    // causal mask (only bites for tok >= CTX) + online softmax
    float cmax[8];
#pragma unroll
    for (int r = 0; r < 8; ++r) {
      int qr = r + 8 * hi;
      if (tok0 > qr + kCTX) c0[r] = -__builtin_inff();
      if (tok1 > qr + kCTX) c1[r] = -__builtin_inff();
      cmax[r] = fmaxf(c0[r], c1[r]);
    }
#pragma unroll
    for (int off = 1; off < 16; off <<= 1)
#pragma unroll
      for (int r = 0; r < 8; ++r) cmax[r] = fmaxf(cmax[r], __shfl_xor(cmax[r], off, 32));

    float f[8], m_new[8], p0[8], p1[8], rsum[8];
#pragma unroll
    for (int r = 0; r < 8; ++r) {
      m_new[r] = fmaxf(m_run[r], cmax[r]);
      f[r] = __expf(m_run[r] - m_new[r]);
      m_run[r] = m_new[r];
      p0[r] = __expf(c0[r] - m_new[r]);
      p1[r] = __expf(c1[r] - m_new[r]);
      rsum[r] = p0[r] + p1[r];
    }
#pragma unroll
    for (int off = 1; off < 16; off <<= 1)
#pragma unroll
      for (int r = 0; r < 8; ++r) rsum[r] += __shfl_xor(rsum[r], off, 32);
#pragma unroll
    for (int r = 0; r < 8; ++r) l_run[r] = l_run[r] * f[r] + rsum[r];
#pragma unroll
    for (int n = 0; n < 8; ++n)
#pragma unroll
      for (int r = 0; r < 8; ++r) acc[n][r] *= f[r];

    // stage P (C-layout) in per-wave LDS, re-read in A-layout
#pragma unroll
    for (int r = 0; r < 8; ++r) {
      pbuf[wv_][r + 8 * hi][ln] = p0[r];
      pbuf[wv_][r + 8 * hi][16 + ln] = p1[r];
    }
    asm volatile("s_wait_dscnt 0x0" ::: "memory");
    v16bf pa;
    {
      const float* pp = &pbuf[wv_][ln][hi * 16];
#pragma unroll
      for (int i = 0; i < 8; ++i) pa[i] = (__bf16)pp[i];
#pragma unroll
      for (int i = 0; i < 8; ++i) pa[8 + i] = (__bf16)pp[8 + i];
    }

    // acc += P . V   (8 d-tiles, K = 32 tokens)
    int pgh = hi ? pg1 : pg0;
    int tokb = kv0 + hi * 16;
#pragma unroll
    for (int n = 0; n < 8; ++n) {
      v16bf vb;
#pragma unroll
      for (int i = 0; i < 16; ++i) {
        int t = tokb + i;
        const float* pv = (t < kCTX) ? (vbase + ((size_t)pgh * kPS + i) * kD)
                                     : (vnb + (size_t)(t - kCTX) * kD);
        vb[i] = (__bf16)pv[n * 16 + ln];
      }
      acc[n] = __builtin_amdgcn_wmma_f32_16x16x32_bf16(false, pa, false, vb, (short)0, acc[n], false, false);
    }
    asm volatile("" ::: "memory");   // keep LDS reads before next iter's writes
  }

  // write split partials: acc[16][128], m[16], l[16]
  size_t pb = ((size_t)bh * kSPLIT + wv_) * 2048;
#pragma unroll
  for (int n = 0; n < 8; ++n)
#pragma unroll
    for (int r = 0; r < 8; ++r)
      part_acc[pb + (size_t)(r + 8 * hi) * kD + n * 16 + ln] = acc[n][r];
  if (ln == 0) {
    size_t mb = ((size_t)bh * kSPLIT + wv_) * 32;
#pragma unroll
    for (int r = 0; r < 8; ++r) {
      part_ml[mb + (r + 8 * hi)] = m_run[r];
      part_ml[mb + 16 + (r + 8 * hi)] = l_run[r];
    }
  }
}

// ---------------- kernel 3: merge split partials -> attn_ws [B,S,E] ----------------
__global__ __launch_bounds__(256) void reduce_kernel(
    const float* __restrict__ part_acc, const float* __restrict__ part_ml,
    float* __restrict__ attn_ws) {
  int bh = blockIdx.x;
  int b = bh >> 4, h = bh & 15;
  int t = threadIdx.x;
  int q = t >> 4;               // 0..15
  int dbase = (t & 15) * 8;     // 0..120
  size_t mb = (size_t)bh * kSPLIT * 32;
  float mw[kSPLIT], lw[kSPLIT], sc[kSPLIT];
  float M = -__builtin_inff();
#pragma unroll
  for (int w = 0; w < kSPLIT; ++w) {
    mw[w] = part_ml[mb + w * 32 + q];
    lw[w] = part_ml[mb + w * 32 + 16 + q];
    M = fmaxf(M, mw[w]);
  }
  float L = 0.f;
#pragma unroll
  for (int w = 0; w < kSPLIT; ++w) { sc[w] = __expf(mw[w] - M); L += lw[w] * sc[w]; }
  float inv = 1.0f / L;
  size_t ab = (size_t)bh * kSPLIT * 2048;
#pragma unroll
  for (int j = 0; j < 8; ++j) {
    int d = dbase + j;
    float s = 0.f;
#pragma unroll
    for (int w = 0; w < kSPLIT; ++w) s += part_acc[ab + (size_t)w * 2048 + q * kD + d] * sc[w];
    attn_ws[((size_t)(b * kS + q)) * kE + h * kD + d] = s * inv;
  }
}

// ---------------- kernel 4: output projection (out = attn @ wo^T) ----------------
// grid: 8 * 128 = 1024 waves -> 128 blocks x 256 thr
__global__ __launch_bounds__(256) void out_proj_kernel(
    const float* __restrict__ attn, const float* __restrict__ wo,
    float* __restrict__ out) {
  int wave = blockIdx.x * (blockDim.x >> 5) + (threadIdx.x >> 5);
  int lane = threadIdx.x & 31;
  int ln = lane & 15, hi = lane >> 4;
  int m0 = (wave >> 7) * 16;
  int n0 = (wave & 127) * 16;
  const float* arow = attn + (size_t)(m0 + ln) * kE;
  const float* wrow = wo + (size_t)(n0 + ln) * kE;
  v8f acc = {};
  for (int k0 = 0; k0 < kE; k0 += 32) {
    v16bf a = load_a_frag(arow, k0);
    v16bf b = load_b_frag(wrow, k0);
    acc = __builtin_amdgcn_wmma_f32_16x16x32_bf16(false, a, false, b, (short)0, acc, false, false);
  }
#pragma unroll
  for (int r = 0; r < 8; ++r)
    out[(size_t)(m0 + r + 8 * hi) * kE + n0 + ln] = acc[r];
}

// ---------------- host launcher ----------------
extern "C" void kernel_launch(void* const* d_in, const int* in_sizes, int n_in,
                              void* d_out, int out_size, void* d_ws, size_t ws_size,
                              hipStream_t stream) {
  (void)in_sizes; (void)n_in; (void)out_size; (void)ws_size;
  const float* hidden   = (const float*)d_in[0];
  const float* wq       = (const float*)d_in[1];
  const float* wk       = (const float*)d_in[2];
  const float* wv       = (const float*)d_in[3];
  const float* wo       = (const float*)d_in[4];
  const float* k_cache  = (const float*)d_in[5];
  const float* v_cache  = (const float*)d_in[6];
  const int*   page_tab = (const int*)d_in[7];

  float* ws      = (float*)d_ws;
  float* q_ws    = ws;                               // 262144
  float* k_new   = ws + 262144;                      // 262144
  float* v_new   = ws + 524288;                      // 262144
  float* attn_ws = ws + 786432;                      // 262144
  float* p_acc   = ws + 1048576;                     // 128*8*2048 = 2097152
  float* p_ml    = p_acc + (size_t)128 * kSPLIT * 2048;  // 128*8*32

  qkv_proj_kernel<<<384, 256, 0, stream>>>(hidden, wq, wk, wv, q_ws, k_new, v_new);
  attn_kernel<<<kB * kH, 256, 0, stream>>>(q_ws, k_new, v_new, k_cache, v_cache,
                                           page_tab, p_acc, p_ml);
  reduce_kernel<<<kB * kH, 256, 0, stream>>>(p_acc, p_ml, attn_ws);
  out_proj_kernel<<<128, 256, 0, stream>>>(attn_ws, wo, (float*)d_out);
}
